// PainnModel_57200374448313
// MI455X (gfx1250) — matrix operand order repaired and checked by reading, not compiled, not run on hardware
//
#include <hip/hip_runtime.h>
#include <hip/hip_bf16.h>
#include <math.h>

#define NN 10000
#define EE 160000
#define FF 128
#define LL 3
#define PI_F 3.14159265358979f

typedef __bf16 bf16;
typedef __bf16 v16bf __attribute__((ext_vector_type(16)));
typedef float  v8f   __attribute__((ext_vector_type(8)));

// ---------------------------------------------------------------------------
// Generic WMMA GEMM:  C[M x Nc] = act( A[M x K] @ W[K x Nc] + bias[Nc] )
// A: fp32 row-major.  WT: bf16, transposed (Nc x K row-major), K mult of 32.
// One wave computes one 16x16 tile; 4 waves / block; grid = (Nc/16, ceil(M/64)).
// act: 0 = identity, 1 = SiLU.
// ---------------------------------------------------------------------------
__device__ inline v16bf load_a_frag(const float* A, int row, int K, int k0, int kg) {
    const float* p0 = A + (size_t)row * K + k0 + 16 * kg;       // K = {0..7}+16kg
    const float* p1 = p0 + 8;                                   // K = {8..15}+16kg
    float4 x0 = *(const float4*)(p0);
    float4 x1 = *(const float4*)(p0 + 4);
    float4 x2 = *(const float4*)(p1);
    float4 x3 = *(const float4*)(p1 + 4);
    v16bf r;
    r[0]  = (bf16)x0.x; r[1]  = (bf16)x0.y; r[2]  = (bf16)x0.z; r[3]  = (bf16)x0.w;
    r[4]  = (bf16)x1.x; r[5]  = (bf16)x1.y; r[6]  = (bf16)x1.z; r[7]  = (bf16)x1.w;
    r[8]  = (bf16)x2.x; r[9]  = (bf16)x2.y; r[10] = (bf16)x2.z; r[11] = (bf16)x2.w;
    r[12] = (bf16)x3.x; r[13] = (bf16)x3.y; r[14] = (bf16)x3.z; r[15] = (bf16)x3.w;
    return r;
}

__global__ void wmma_gemm_bias_act(const float* __restrict__ A,
                                   const bf16*  __restrict__ WT,
                                   const float* __restrict__ bias,
                                   float* __restrict__ C,
                                   int M, int K, int Nc, int act) {
    const int lane = threadIdx.x & 31;
    const int wave = threadIdx.x >> 5;
    const int kg   = lane >> 4;          // K-group within fragment layouts
    const int rt   = blockIdx.y * 4 + wave;          // 16-row tile index
    const int cn   = blockIdx.x * 16;                // 16-col tile base
    const int r16  = rt * 16;

    int rowA = r16 + (lane & 15);
    if (rowA > M - 1) rowA = M - 1;                  // clamp loads; stores masked
    const int col = cn + (lane & 15);

    v8f acc = {0.f, 0.f, 0.f, 0.f, 0.f, 0.f, 0.f, 0.f};
    for (int k0 = 0; k0 < K; k0 += 32) {
        v16bf af = load_a_frag(A, rowA, K, k0, kg);
        // B fragment: lane (n + 16g) holds column n, K = k0 + 16g .. +15 (contiguous)
        v16bf bf = *(const v16bf*)(WT + (size_t)col * K + k0 + 16 * kg);
        acc = __builtin_amdgcn_wmma_f32_16x16x32_bf16(false, af, false, bf,
                                                      (short)0, acc, false, false);
    }

    const float bv = bias[col];
    const int rbase = r16 + 8 * kg;                  // D layout: M = i + 8*kg
#pragma unroll
    for (int i = 0; i < 8; ++i) {
        int r = rbase + i;
        if (r < M) {
            float v = acc[i] + bv;
            if (act) v = v / (1.f + __expf(-v)) ;    // SiLU
            C[(size_t)r * Nc + col] = v;
        }
    }
}

// ---------------------------------------------------------------------------
// Weight preprocessing: W[K x Nc] fp32 -> WT[Nc x Kpad] bf16 (zero-padded K)
// ---------------------------------------------------------------------------
__global__ void transpose_cast_k(const float* __restrict__ W, bf16* __restrict__ WT,
                                 int K, int Nc, int Kpad) {
    int idx = blockIdx.x * blockDim.x + threadIdx.x;
    int total = Nc * Kpad;
    if (idx >= total) return;
    int n = idx / Kpad, k = idx - n * Kpad;
    WT[idx] = (k < K) ? (bf16)W[(size_t)k * Nc + n] : (bf16)0.f;
}

// ---------------------------------------------------------------------------
// Edge geometry: rbf (padded to 32), envelope, unit vectors
// ---------------------------------------------------------------------------
__global__ void edge_precompute(const float* __restrict__ diff,
                                float* __restrict__ rbf32,
                                float* __restrict__ env,
                                float* __restrict__ unitv, int E) {
    int e = blockIdx.x * blockDim.x + threadIdx.x;
    if (e >= E) return;
    float dx = diff[3 * e], dy = diff[3 * e + 1], dz = diff[3 * e + 2];
    float d = sqrtf(dx * dx + dy * dy + dz * dz);
    float inv = 1.f / d;
    float* r = rbf32 + (size_t)e * 32;
#pragma unroll
    for (int j = 0; j < 20; ++j) r[j] = sinf(d * (float)(j + 1) * (PI_F / 5.f)) * inv;
#pragma unroll
    for (int j = 20; j < 32; ++j) r[j] = 0.f;
    env[e] = (d < 5.f) ? 0.5f * (cosf(PI_F * d * 0.2f) + 1.f) : 0.f;
    unitv[3 * e] = dx * inv; unitv[3 * e + 1] = dy * inv; unitv[3 * e + 2] = dz * inv;
}

// ns = embed[elems], nv = 0
__global__ void node_init(const int* __restrict__ elems, const float* __restrict__ embed,
                          float* __restrict__ ns, float* __restrict__ nv, int N) {
    int idx = blockIdx.x * blockDim.x + threadIdx.x;
    if (idx >= N * 384) return;
    int n = idx / 384, j = idx - n * 384;
    nv[idx] = 0.f;
    if (j < FF) ns[n * FF + j] = embed[(size_t)elems[n] * FF + j];
}

__global__ void copy_f32(const float* __restrict__ src, float* __restrict__ dst, int n) {
    for (int i = blockIdx.x * blockDim.x + threadIdx.x; i < n; i += gridDim.x * blockDim.x)
        dst[i] = src[i];
}

// ---------------------------------------------------------------------------
// Edge message kernel (filter GEMM K=20 fused): one 128-thread block per edge.
//   fw = (rbf @ filtW + b) * env ; fo = fw * s[src] ; scatter-add to dst.
// ---------------------------------------------------------------------------
__global__ void edge_msg(const int* __restrict__ pairs,
                         const float* __restrict__ rbf32,
                         const float* __restrict__ env,
                         const float* __restrict__ unitv,
                         const bf16*  __restrict__ filtT,   // [384 x 32]
                         const float* __restrict__ filt_b,  // [384]
                         const float* __restrict__ s,       // [N x 384]
                         const float* __restrict__ nv_old,  // [N x 3 x 128]
                         float* __restrict__ ns_new,
                         float* __restrict__ nv_new) {
    const int e = blockIdx.x;
    const int f = threadIdx.x;                       // feature 0..127
    __shared__ float rs[32];
    if (f < 32) rs[f] = rbf32[(size_t)e * 32 + f];
    __syncthreads();

    const int dst = pairs[2 * e], src = pairs[2 * e + 1];
    const float envv = env[e];
    float fw[3];
#pragma unroll
    for (int g = 0; g < 3; ++g) {
        const int c = f + 128 * g;
        const bf16* w = filtT + (size_t)c * 32;
        float a = 0.f;
#pragma unroll
        for (int k = 0; k < 32; ++k) a += rs[k] * (float)w[k];
        fw[g] = (a + filt_b[c]) * envv;
    }
    const float* sp = s + (size_t)src * 384;
    const float gsv = fw[0] * sp[f];
    const float gev = fw[1] * sp[128 + f];
    const float ms  = fw[2] * sp[256 + f];
    atomicAdd(ns_new + (size_t)dst * FF + f, ms);
    const float* nvp = nv_old + (size_t)src * 384;
    float* nvo = nv_new + (size_t)dst * 384;
#pragma unroll
    for (int d = 0; d < 3; ++d) {
        float mv = nvp[d * FF + f] * gsv + gev * unitv[(size_t)e * 3 + d];
        atomicAdd(nvo + d * FF + f, mv);
    }
}

// h = [ ||Vv||_spatial , ns ]   (N x 256)
__global__ void vn_concat(const float* __restrict__ Vv, const float* __restrict__ ns,
                          float* __restrict__ h, int N) {
    int idx = blockIdx.x * blockDim.x + threadIdx.x;
    if (idx >= N * FF) return;
    int n = idx >> 7, f = idx & 127;
    const float* vp = Vv + (size_t)n * 384;
    float v0 = vp[f], v1 = vp[128 + f], v2 = vp[256 + f];
    h[(size_t)n * 256 + f]       = sqrtf(v0 * v0 + v1 * v1 + v2 * v2);
    h[(size_t)n * 256 + 128 + f] = ns[(size_t)n * FF + f];
}

// ns += a_sv * <Uv,Vv> + a_ss ; nv += a_vv * Uv
__global__ void apply_update(const float* __restrict__ o, const float* __restrict__ Uv,
                             const float* __restrict__ Vv, float* __restrict__ ns,
                             float* __restrict__ nv, int N) {
    int idx = blockIdx.x * blockDim.x + threadIdx.x;
    if (idx >= N * FF) return;
    int n = idx >> 7, f = idx & 127;
    const float* up = Uv + (size_t)n * 384;
    const float* vp = Vv + (size_t)n * 384;
    const float* op = o  + (size_t)n * 384;
    float a_vv = op[f], a_sv = op[128 + f], a_ss = op[256 + f];
    float inner = up[f] * vp[f] + up[128 + f] * vp[128 + f] + up[256 + f] * vp[256 + f];
    ns[(size_t)n * FF + f] += a_sv * inner + a_ss;
    float* nvp = nv + (size_t)n * 384;
#pragma unroll
    for (int d = 0; d < 3; ++d) nvp[d * FF + f] += a_vv * up[d * FF + f];
}

__global__ void zero1(float* out) { out[0] = 0.f; }

// sum_n ( r1[n] . w2 + b2 )  -> out[0]
__global__ void readout_sum(const float* __restrict__ r1, const float* __restrict__ w2,
                            const float* __restrict__ b2, float* __restrict__ out) {
    const int n = blockIdx.x, f = threadIdx.x;
    __shared__ float red[128];
    red[f] = r1[(size_t)n * FF + f] * w2[f];
    __syncthreads();
    for (int s = 64; s > 0; s >>= 1) {
        if (f < s) red[f] += red[f + s];
        __syncthreads();
    }
    if (f == 0) atomicAdd(out, red[0] + b2[0]);
}

// ---------------------------------------------------------------------------
extern "C" void kernel_launch(void* const* d_in, const int* in_sizes, int n_in,
                              void* d_out, int out_size, void* d_ws, size_t ws_size,
                              hipStream_t stream) {
    const int N = NN, E = EE;

    const int*   elems      = (const int*)  d_in[0];
    const int*   pairs      = (const int*)  d_in[1];
    const float* edge_diff  = (const float*)d_in[2];
    const float* embed      = (const float*)d_in[3];
    const float* mfW        = (const float*)d_in[4];   // (L,20,384)
    const float* mfB        = (const float*)d_in[5];   // (L,384)
    const float* m1W        = (const float*)d_in[6];   // (L,128,128)
    const float* m1B        = (const float*)d_in[7];
    const float* m2W        = (const float*)d_in[8];   // (L,128,384)
    const float* m2B        = (const float*)d_in[9];
    const float* uUW        = (const float*)d_in[10];  // (L,128,128)
    const float* uUB        = (const float*)d_in[11];
    const float* uVW        = (const float*)d_in[12];
    const float* uVB        = (const float*)d_in[13];
    const float* u1W        = (const float*)d_in[14];  // (L,256,128)
    const float* u1B        = (const float*)d_in[15];
    const float* u2W        = (const float*)d_in[16];  // (L,128,384)
    const float* u2B        = (const float*)d_in[17];
    const float* r1W        = (const float*)d_in[18];  // (128,128)
    const float* r1B        = (const float*)d_in[19];
    const float* r2W        = (const float*)d_in[20];  // (128,1)
    const float* r2B        = (const float*)d_in[21];
    float* out = (float*)d_out;

    // ---- workspace arena (256B-aligned carves) ----
    char* base = (char*)d_ws;
    size_t off = 0;
    auto carve = [&](size_t bytes) -> void* {
        void* p = base + off;
        off += (bytes + 255) & ~((size_t)255);
        return p;
    };
    float* rbf32 = (float*)carve((size_t)E * 32 * 4);
    float* envp  = (float*)carve((size_t)E * 4);
    float* unitp = (float*)carve((size_t)E * 3 * 4);
    float* nsb[2], *nvb[2];
    nsb[0] = (float*)carve((size_t)N * 128 * 4);
    nsb[1] = (float*)carve((size_t)N * 128 * 4);
    nvb[0] = (float*)carve((size_t)N * 384 * 4);
    nvb[1] = (float*)carve((size_t)N * 384 * 4);
    float* s    = (float*)carve((size_t)N * 384 * 4);
    float* h1   = (float*)carve((size_t)N * 128 * 4);
    float* Uv   = (float*)carve((size_t)N * 384 * 4);
    float* Vv   = (float*)carve((size_t)N * 384 * 4);
    float* hcat = (float*)carve((size_t)N * 256 * 4);
    float* t1   = (float*)carve((size_t)N * 128 * 4);
    float* ovec = (float*)carve((size_t)N * 384 * 4);
    float* r1   = (float*)carve((size_t)N * 128 * 4);
    bf16* filtT = (bf16*)carve((size_t)LL * 384 * 32 * 2);
    bf16* m1T   = (bf16*)carve((size_t)LL * 128 * 128 * 2);
    bf16* m2T   = (bf16*)carve((size_t)LL * 384 * 128 * 2);
    bf16* uUT   = (bf16*)carve((size_t)LL * 128 * 128 * 2);
    bf16* uVT   = (bf16*)carve((size_t)LL * 128 * 128 * 2);
    bf16* u1T   = (bf16*)carve((size_t)LL * 128 * 256 * 2);
    bf16* u2T   = (bf16*)carve((size_t)LL * 384 * 128 * 2);
    bf16* roT   = (bf16*)carve((size_t)128 * 128 * 2);
    if (off > ws_size) return;  // workspace too small: bail (no-op)

    auto tc = [&](const float* W, bf16* WT, int K, int Nc, int Kpad) {
        int tot = Nc * Kpad;
        transpose_cast_k<<<(tot + 255) / 256, 256, 0, stream>>>(W, WT, K, Nc, Kpad);
    };
    for (int l = 0; l < LL; ++l) {
        tc(mfW + (size_t)l * 20 * 384,  filtT + (size_t)l * 384 * 32,  20, 384, 32);
        tc(m1W + (size_t)l * 128 * 128, m1T   + (size_t)l * 128 * 128, 128, 128, 128);
        tc(m2W + (size_t)l * 128 * 384, m2T   + (size_t)l * 384 * 128, 128, 384, 128);
        tc(uUW + (size_t)l * 128 * 128, uUT   + (size_t)l * 128 * 128, 128, 128, 128);
        tc(uVW + (size_t)l * 128 * 128, uVT   + (size_t)l * 128 * 128, 128, 128, 128);
        tc(u1W + (size_t)l * 256 * 128, u1T   + (size_t)l * 128 * 256, 256, 128, 256);
        tc(u2W + (size_t)l * 128 * 384, u2T   + (size_t)l * 384 * 128, 128, 384, 128);
    }
    tc(r1W, roT, 128, 128, 128);

    edge_precompute<<<(E + 255) / 256, 256, 0, stream>>>(edge_diff, rbf32, envp, unitp, E);
    node_init<<<(N * 384 + 255) / 256, 256, 0, stream>>>(elems, embed, nsb[0], nvb[0], N);

    auto gemm = [&](const float* A, const bf16* WT, const float* b, float* C,
                    int M, int K, int Nc, int act) {
        dim3 grid(Nc / 16, (M + 63) / 64);
        wmma_gemm_bias_act<<<grid, 128, 0, stream>>>(A, WT, b, C, M, K, Nc, act);
    };

    int cur = 0;
    for (int l = 0; l < LL; ++l) {
        int nxt = cur ^ 1;
        // s = silu(ns@W1+b1)@W2+b2
        gemm(nsb[cur], m1T + (size_t)l * 128 * 128, m1B + (size_t)l * 128, h1, N, 128, 128, 1);
        gemm(h1,       m2T + (size_t)l * 384 * 128, m2B + (size_t)l * 384, s,  N, 128, 384, 0);
        // accumulate messages into copies of ns/nv
        copy_f32<<<2048, 256, 0, stream>>>(nsb[cur], nsb[nxt], N * 128);
        copy_f32<<<2048, 256, 0, stream>>>(nvb[cur], nvb[nxt], N * 384);
        edge_msg<<<E, 128, 0, stream>>>(pairs, rbf32, envp, unitp,
                                        filtT + (size_t)l * 384 * 32,
                                        mfB + (size_t)l * 384, s, nvb[cur],
                                        nsb[nxt], nvb[nxt]);
        // update block
        gemm(nvb[nxt], uUT + (size_t)l * 128 * 128, uUB + (size_t)l * 128, Uv, 3 * N, 128, 128, 0);
        gemm(nvb[nxt], uVT + (size_t)l * 128 * 128, uVB + (size_t)l * 128, Vv, 3 * N, 128, 128, 0);
        vn_concat<<<(N * 128 + 255) / 256, 256, 0, stream>>>(Vv, nsb[nxt], hcat, N);
        gemm(hcat, u1T + (size_t)l * 128 * 256, u1B + (size_t)l * 128, t1,   N, 256, 128, 1);
        gemm(t1,   u2T + (size_t)l * 384 * 128, u2B + (size_t)l * 384, ovec, N, 128, 384, 0);
        apply_update<<<(N * 128 + 255) / 256, 256, 0, stream>>>(ovec, Uv, Vv,
                                                                nsb[nxt], nvb[nxt], N);
        cur = nxt;
    }

    // readout
    gemm(nsb[cur], roT, r1B, r1, N, 128, 128, 1);
    zero1<<<1, 1, 0, stream>>>(out);
    readout_sum<<<N, 128, 0, stream>>>(r1, r2W, r2B, out);
}
